// ResidualCatastropheProgramLibrary_43224550867476
// MI455X (gfx1250) — matrix operand order repaired and checked by reading, not compiled, not run on hardware
//
#include <hip/hip_runtime.h>

// ---------------------------------------------------------------------------
// MI455X (gfx1250, wave32) fused kernel for the E8 residual-quantizer program.
// Compute-bound (~11 GFLOP matmul work vs ~75 MB traffic -> ~3us @ 23.3TB/s),
// so: all matmuls on v_wmma_f32_16x16x32_f16, analytic O(8) E8 nearest-root
// instead of 240-way cdist, weights staged in LDS (236KB of the 320KB/WGP).
// v2: __launch_bounds__(256,1) + kept[] -> global RMW to kill scratch spills;
//     fused 2-tile wave loop so B fragments load once per K-chunk; softmax
//     logits computed once and stored as packed f16 pairs.
// ---------------------------------------------------------------------------

typedef _Float16 v16h __attribute__((ext_vector_type(16)));
typedef _Float16 v8h  __attribute__((ext_vector_type(8)));
typedef float    v8f  __attribute__((ext_vector_type(8)));

#define SHUF16(lo, hi) __builtin_shufflevector((lo), (hi), 0,1,2,3,4,5,6,7,8,9,10,11,12,13,14,15)

__device__ __forceinline__ float gelu_exact(float x) {
  return 0.5f * x * (1.0f + erff(x * 0.70710678118654752f));
}

union HPack { unsigned int u; _Float16 h[2]; };

// LDS layout (bytes, all 16B aligned). Strides padded for LDS bank spread and
// so every v8h (16B) fragment load is 16B aligned.
#define ROWS        256
#define ATTN_STR    264   // halves per attention row (K padded 240->256, +8 pad)
#define C0_STR      264   // halves per column of [emb0 | ctrl | 0] (K dim)
#define W_STR       72    // halves per column of refine W^T (K 52->64, +8 pad)
#define EMB_STR     72    // halves per row of emb/h buffer (N 52->64, +8 pad)

#define OFF_ATTN    0                         // 256*264*2 = 135168
#define OFF_C0      135168                    // 64*264*2  =  33792
#define OFF_W1      168960                    // 64*72*2   =   9216
#define OFF_W2      178176                    //              9216
#define OFF_EMBH    187392                    // 256*72*2  =  36864
#define OFF_ROOTS   224256                    // 240*8*4   =   7680
#define OFF_CTRL    231936                    // 256*4*4   =   4096
#define OFF_RSUM    236032                    // 256*4     =   1024
#define OFF_BCL     237056                    // 240*4     ->pad 960
#define OFF_PW1     238016                    // 448*4     =   1792
#define OFF_PB1     239808                    // 32*4
#define OFF_PW2     239936                    // 256*4
#define OFF_PB2     240960                    // 8*4
#define OFF_RB1     240992                    // 64*4
#define OFF_RB2     241248                    // 64*4
#define SMEM_BYTES  241536

__global__ __launch_bounds__(256, 1)
void ResidualCatastropheProgramLibrary_43224550867476_kernel(
    const float* __restrict__ obs,    // [B,14]
    const float* __restrict__ pw1g,   // [32,14]
    const float* __restrict__ pb1g,   // [32]
    const float* __restrict__ pw2g,   // [8,32]
    const float* __restrict__ pb2g,   // [8]
    const float* __restrict__ lvlE,   // [8,240,52]
    const float* __restrict__ bctl,   // [240,4]
    const float* __restrict__ rctl,   // [7,240,4]
    const float* __restrict__ bcg,    // [240]
    const float* __restrict__ logd,   // [1]
    const float* __restrict__ rw1,    // [52,52]
    const float* __restrict__ rb1g,   // [52]
    const float* __restrict__ rw2,    // [52,52]
    const float* __restrict__ rb2g,   // [52]
    const float* __restrict__ rgate,  // [1]
    float* __restrict__ out)          // [B,56]
{
  extern __shared__ char smem[];
  _Float16* attn   = (_Float16*)(smem + OFF_ATTN);
  _Float16* c0T    = (_Float16*)(smem + OFF_C0);   // [n=0..63][k=0..263]: emb0|ctrl|0, transposed
  _Float16* w1T    = (_Float16*)(smem + OFF_W1);   // [n][k] = refine_w1[n,k]
  _Float16* w2T    = (_Float16*)(smem + OFF_W2);
  _Float16* embH   = (_Float16*)(smem + OFF_EMBH); // [row][64] f16 emb / hidden buffer
  float*    rootsN = (float*)(smem + OFF_ROOTS);   // roots / sqrt(2)
  float*    ctrlB  = (float*)(smem + OFF_CTRL);
  float*    rsumI  = (float*)(smem + OFF_RSUM);
  float*    bclL   = (float*)(smem + OFF_BCL);     // base_complexity * ln2
  float*    pw1    = (float*)(smem + OFF_PW1);
  float*    pb1    = (float*)(smem + OFF_PB1);
  float*    pw2    = (float*)(smem + OFF_PW2);
  float*    pb2    = (float*)(smem + OFF_PB2);
  float*    rb1L   = (float*)(smem + OFF_RB1);
  float*    rb2L   = (float*)(smem + OFF_RB2);

  const int tid = threadIdx.x;

  // ---------------- Stage weights / tables into LDS ----------------
  for (int i = tid; i < 64 * C0_STR; i += 256) {
    int n = i / C0_STR, k = i % C0_STR;
    float v = 0.f;
    if (k < 240) {
      if (n < 52)      v = lvlE[k * 52 + n];          // level-0 embeddings
      else if (n < 56) v = bctl[k * 4 + (n - 52)];    // base control packed as cols 52..55
    }
    c0T[n * C0_STR + k] = (_Float16)v;
  }
  for (int i = tid; i < 64 * W_STR; i += 256) {
    int n = i / W_STR, k = i % W_STR;
    float a = (n < 52 && k < 52) ? rw1[n * 52 + k] : 0.f;
    float b = (n < 52 && k < 52) ? rw2[n * 52 + k] : 0.f;
    w1T[n * W_STR + k] = (_Float16)a;
    w2T[n * W_STR + k] = (_Float16)b;
  }
  if (tid < 240) {
    // Rebuild the E8 root table in reference enumeration order.
    float rt[8] = {0.f,0.f,0.f,0.f,0.f,0.f,0.f,0.f};
    if (tid < 112) {
      int p = tid >> 2, i = 0, acc = 0;
      while (p >= acc + (7 - i)) { acc += 7 - i; ++i; }
      int j = i + 1 + (p - acc);
      rt[i] = (tid & 2) ? -1.f : 1.f;
      rt[j] = (tid & 1) ? -1.f : 1.f;
    } else {
      int bits = (tid - 112) << 1;
      if (__popc(bits) & 1) bits |= 1;   // pick the even-parity member of the pair
      #pragma unroll
      for (int k = 0; k < 8; ++k) rt[k] = ((bits >> k) & 1) ? -0.5f : 0.5f;
    }
    #pragma unroll
    for (int k = 0; k < 8; ++k) rootsN[tid * 8 + k] = rt[k] * 0.70710678118654752f;
    bclL[tid] = bcg[tid] * 0.69314718055994531f;
  }
  for (int i = tid; i < 448; i += 256) pw1[i] = pw1g[i];
  pw2[tid] = pw2g[tid];                       // 256 entries
  if (tid < 32) pb1[tid] = pb1g[tid];
  if (tid < 8)  pb2[tid] = pb2g[tid];
  if (tid < 64) {
    rb1L[tid] = (tid < 52) ? rb1g[tid] : 0.f;
    rb2L[tid] = (tid < 52) ? rb2g[tid] : 0.f;
  }
  __syncthreads();

  // ---------------- Phase A: per-row VALU/trans work ----------------
  {
    const int row = tid;
    const long gRow = (long)blockIdx.x * ROWS + row;
    unsigned int* attnU = (unsigned int*)(attn + row * ATTN_STR);

    float ob[14];
    #pragma unroll
    for (int k = 0; k < 14; ++k) ob[k] = obs[gRow * 14 + k];

    float h[32];
    #pragma unroll
    for (int o = 0; o < 32; ++o) {
      float s = pb1[o];
      #pragma unroll
      for (int k = 0; k < 14; ++k) s = fmaf(pw1[o * 14 + k], ob[k], s);
      h[o] = gelu_exact(s);
    }
    float q[8];
    #pragma unroll
    for (int p = 0; p < 8; ++p) {
      float s = pb2[p];
      #pragma unroll
      for (int k = 0; k < 32; ++k) s = fmaf(pw2[p * 32 + k], h[k], s);
      q[p] = s;
    }
    float nn = 0.f;
    #pragma unroll
    for (int p = 0; p < 8; ++p) nn = fmaf(q[p], q[p], nn);
    float qs = 1.41421356237f / fmaxf(sqrtf(nn), 1e-12f);
    #pragma unroll
    for (int p = 0; p < 8; ++p) q[p] *= qs;

    // Softmax over 240 logits. Single dot pass: store logits as packed f16
    // pairs while tracking the max; exp pass re-reads them. 1/sum is folded
    // into the WMMA epilogue (softmax normalization commutes with the matmul).
    float mx = -1e30f;
    for (int r = 0; r < 240; r += 2) {
      const float* rp = rootsN + r * 8;
      float s0 = -bclL[r], s1 = -bclL[r + 1];
      #pragma unroll
      for (int k = 0; k < 8; ++k) {
        s0 = fmaf(q[k], rp[k], s0);
        s1 = fmaf(q[k], rp[k + 8], s1);
      }
      mx = fmaxf(mx, fmaxf(s0, s1));
      HPack p2; p2.h[0] = (_Float16)s0; p2.h[1] = (_Float16)s1;
      attnU[r >> 1] = p2.u;
    }
    float sum = 0.f;
    for (int r = 0; r < 240; r += 2) {
      HPack p2; p2.u = attnU[r >> 1];
      float e0 = __expf((float)p2.h[0] - mx);
      float e1 = __expf((float)p2.h[1] - mx);
      sum += e0 + e1;
      p2.h[0] = (_Float16)e0; p2.h[1] = (_Float16)e1;
      attnU[r >> 1] = p2.u;
    }
    #pragma unroll
    for (int r = 240; r < 256; r += 2) attnU[r >> 1] = 0u;
    rsumI[row] = 1.f / sum;

    // Analytic residual E8 quantization: argmin||.|| == argmax<residual,root>
    // (all roots have norm^2 == 2). O(8) per level instead of 240x8.
    float decay = __expf(logd[0]);
    float invd  = 1.f / decay;
    float res[8];
    #pragma unroll
    for (int k = 0; k < 8; ++k) res[k] = q[k];

    float gs[52];
    #pragma unroll
    for (int c = 0; c < 52; ++c) gs[c] = 0.f;
    float gc0 = 0.f, gc1 = 0.f, gc2 = 0.f, gc3 = 0.f;

    float scale = 2.f, invl = 1.f;
    for (int l = 0; l < 8; ++l) {
      float a1 = -1.f, a2 = -1.f, sumabs = 0.f, mn = 1e30f;
      int i1 = 0, i2 = 0, im = 0, bits = 0;
      #pragma unroll
      for (int k = 0; k < 8; ++k) {
        float a = fabsf(res[k]);
        sumabs += a;
        if (a < mn) { mn = a; im = k; }
        if (res[k] < 0.f) bits |= (1 << k);
        if (a > a1)      { a2 = a1; i2 = i1; a1 = a; i1 = k; }
        else if (a > a2) { a2 = a; i2 = k; }
      }
      float best1 = a1 + a2;
      int i = (i1 < i2) ? i1 : i2, j = (i1 < i2) ? i2 : i1;
      int b2 = bits;
      float v2;
      if (__popc(bits) & 1) { b2 ^= (1 << im); v2 = 0.5f * (sumabs - 2.f * mn); }
      else                  { v2 = 0.5f * sumabs; }
      int idx;
      if (best1 >= v2) {                       // (±1,±1,0^6) family, ties -> lower index
        int si = (res[i] < 0.f) ? 1 : 0, sj = (res[j] < 0.f) ? 1 : 0;
        int pair = 7 * i - (i * (i - 1)) / 2 + (j - i - 1);
        idx = pair * 4 + si * 2 + sj;
        res[i] -= (si ? -1.f : 1.f) * scale;
        res[j] -= (sj ? -1.f : 1.f) * scale;
      } else {                                 // (±1/2)^8 even-parity family
        idx = 112 + (b2 >> 1);
        #pragma unroll
        for (int k = 0; k < 8; ++k)
          res[k] -= (((b2 >> k) & 1) ? -0.5f : 0.5f) * scale;
      }
      if (l >= 1) {                            // hard-index gathers (L2-resident tables)
        invl *= invd;
        const float4* e4 = (const float4*)(lvlE + ((long)l * 240 + idx) * 52);
        #pragma unroll
        for (int c = 0; c < 13; ++c) {
          float4 v = e4[c];
          gs[c*4+0] = fmaf(v.x, invl, gs[c*4+0]);
          gs[c*4+1] = fmaf(v.y, invl, gs[c*4+1]);
          gs[c*4+2] = fmaf(v.z, invl, gs[c*4+2]);
          gs[c*4+3] = fmaf(v.w, invl, gs[c*4+3]);
        }
        float4 cv = *(const float4*)(rctl + ((long)(l - 1) * 240 + idx) * 4);
        gc0 = fmaf(cv.x, invl, gc0); gc1 = fmaf(cv.y, invl, gc1);
        gc2 = fmaf(cv.z, invl, gc2); gc3 = fmaf(cv.w, invl, gc3);
      }
      scale *= invd;
    }
    #pragma unroll
    for (int c = 0; c < 52; ++c) embH[row * EMB_STR + c] = (_Float16)gs[c];
    #pragma unroll
    for (int c = 52; c < 64; ++c) embH[row * EMB_STR + c] = (_Float16)0.f;
    ctrlB[row * 4 + 0] = gc0; ctrlB[row * 4 + 1] = gc1;
    ctrlB[row * 4 + 2] = gc2; ctrlB[row * 4 + 3] = gc3;
  }
  __syncthreads();

  // -------- Phase B: WMMA, wave owns 32 rows (two fused 16-row tiles) -------
  const int wv = tid >> 5, lane = tid & 31;
  const int half = lane >> 4, l16 = lane & 15;
  const int tb0 = wv * 32;                     // tile t covers rows tb0 + t*16 ..
  const float sgate = 1.f / (1.f + __expf(-rgate[0]));
  const long gBase = (long)blockIdx.x * ROWS;

  // --- attn[32x256] @ [emb0|ctrl][256x64]: B fragments loaded once per kc ---
  v8f acc[2][4] = {};
  #pragma unroll
  for (int kc = 0; kc < 8; ++kc) {
    v16h b[4];
    #pragma unroll
    for (int nt = 0; nt < 4; ++nt) {
      const _Float16* bp = c0T + (nt * 16 + l16) * C0_STR + kc * 32 + 16 * half;
      b[nt] = SHUF16(*(const v8h*)bp, *(const v8h*)(bp + 8));
    }
    #pragma unroll
    for (int t = 0; t < 2; ++t) {
      const _Float16* ap = attn + (tb0 + t * 16 + l16) * ATTN_STR + kc * 32 + 16 * half;
      v16h a = SHUF16(*(const v8h*)ap, *(const v8h*)(ap + 8));
      #pragma unroll
      for (int nt = 0; nt < 4; ++nt)
        acc[t][nt] = __builtin_amdgcn_wmma_f32_16x16x32_f16(false, a, false, b[nt],
                                                            (short)0, acc[t][nt], false, false);
    }
  }

  // Epilogue 1: fold 1/rowsum, add gather-sum, write pre-refine result to
  // global (avoids keeping 64 floats live across the refine matmuls) and
  // seed the f16 refine input in LDS.
  #pragma unroll
  for (int t = 0; t < 2; ++t) {
    #pragma unroll
    for (int nt = 0; nt < 4; ++nt) {
      const int N = nt * 16 + l16;
      #pragma unroll
      for (int r = 0; r < 8; ++r) {
        const int row = tb0 + t * 16 + r + 8 * half;
        float v = acc[t][nt][r] * rsumI[row];
        if (N < 52) {
          float e = v + (float)embH[row * EMB_STR + N];
          embH[row * EMB_STR + N] = (_Float16)e;
          out[(gBase + row) * 56 + N] = e;
        } else if (N < 56) {
          embH[row * EMB_STR + N] = (_Float16)0.f;
          out[(gBase + row) * 56 + N] = v + ctrlB[row * 4 + (N - 52)];
        } else {
          embH[row * EMB_STR + N] = (_Float16)0.f;
        }
      }
    }
  }
  __syncthreads();

  // --- refine layer 1: emb[32x64] @ W1^T[64x64], GELU, back to LDS ---
  v8f hac[2][4] = {};
  #pragma unroll
  for (int kc = 0; kc < 2; ++kc) {
    v16h b[4];
    #pragma unroll
    for (int nt = 0; nt < 4; ++nt) {
      const _Float16* bp = w1T + (nt * 16 + l16) * W_STR + kc * 32 + 16 * half;
      b[nt] = SHUF16(*(const v8h*)bp, *(const v8h*)(bp + 8));
    }
    #pragma unroll
    for (int t = 0; t < 2; ++t) {
      const _Float16* ap = embH + (tb0 + t * 16 + l16) * EMB_STR + kc * 32 + 16 * half;
      v16h a = SHUF16(*(const v8h*)ap, *(const v8h*)(ap + 8));
      #pragma unroll
      for (int nt = 0; nt < 4; ++nt)
        hac[t][nt] = __builtin_amdgcn_wmma_f32_16x16x32_f16(false, a, false, b[nt],
                                                            (short)0, hac[t][nt], false, false);
    }
  }
  #pragma unroll
  for (int t = 0; t < 2; ++t)
    #pragma unroll
    for (int nt = 0; nt < 4; ++nt) {
      const int N = nt * 16 + l16;
      #pragma unroll
      for (int r = 0; r < 8; ++r) {
        const int row = tb0 + t * 16 + r + 8 * half;
        embH[row * EMB_STR + N] = (_Float16)gelu_exact(hac[t][nt][r] + rb1L[N]);
      }
    }
  __syncthreads();

  // --- refine layer 2 + gated residual (global RMW of the L2-hot output) ---
  v8f dac[2][4] = {};
  #pragma unroll
  for (int kc = 0; kc < 2; ++kc) {
    v16h b[4];
    #pragma unroll
    for (int nt = 0; nt < 4; ++nt) {
      const _Float16* bp = w2T + (nt * 16 + l16) * W_STR + kc * 32 + 16 * half;
      b[nt] = SHUF16(*(const v8h*)bp, *(const v8h*)(bp + 8));
    }
    #pragma unroll
    for (int t = 0; t < 2; ++t) {
      const _Float16* ap = embH + (tb0 + t * 16 + l16) * EMB_STR + kc * 32 + 16 * half;
      v16h a = SHUF16(*(const v8h*)ap, *(const v8h*)(ap + 8));
      #pragma unroll
      for (int nt = 0; nt < 4; ++nt)
        dac[t][nt] = __builtin_amdgcn_wmma_f32_16x16x32_f16(false, a, false, b[nt],
                                                            (short)0, dac[t][nt], false, false);
    }
  }
  #pragma unroll
  for (int t = 0; t < 2; ++t)
    #pragma unroll
    for (int nt = 0; nt < 4; ++nt) {
      const int N = nt * 16 + l16;
      if (N < 52) {
        #pragma unroll
        for (int r = 0; r < 8; ++r) {
          const int row = tb0 + t * 16 + r + 8 * half;
          float* op = out + (gBase + row) * 56 + N;
          *op = *op + sgate * (dac[t][nt][r] + rb2L[N]);   // same-wave st->ld ordered
        }
      }
    }
}

extern "C" void kernel_launch(void* const* d_in, const int* in_sizes, int n_in,
                              void* d_out, int out_size, void* d_ws, size_t ws_size,
                              hipStream_t stream) {
  (void)n_in; (void)d_ws; (void)ws_size; (void)out_size;
  const int nRows = in_sizes[0] / 14;           // B = 262144
  const int grid  = (nRows + ROWS - 1) / ROWS;  // B divisible by 256

  (void)hipFuncSetAttribute(
      (const void*)ResidualCatastropheProgramLibrary_43224550867476_kernel,
      hipFuncAttributeMaxDynamicSharedMemorySize, SMEM_BYTES);

  ResidualCatastropheProgramLibrary_43224550867476_kernel<<<grid, 256, SMEM_BYTES, stream>>>(
      (const float*)d_in[0],  (const float*)d_in[1],  (const float*)d_in[2],
      (const float*)d_in[3],  (const float*)d_in[4],  (const float*)d_in[5],
      (const float*)d_in[6],  (const float*)d_in[7],  (const float*)d_in[8],
      (const float*)d_in[9],  (const float*)d_in[10], (const float*)d_in[11],
      (const float*)d_in[12], (const float*)d_in[13], (const float*)d_in[14],
      (float*)d_out);
}
